// user_preference_estimator_33062658244949
// MI455X (gfx1250) — matrix operand order repaired and checked by reading, not compiled, not run on hardware
//
#include <hip/hip_runtime.h>

typedef __attribute__((ext_vector_type(16))) _Float16 v16h;
typedef __attribute__((ext_vector_type(8)))  float    v8f;

#define XCOLS 10246
#define HSTRIDE 256
#define ROWS_PER_BLOCK 64
#define NTHREADS 128   // 4 waves (wave32) * 16 rows each

#define K_GENRE    25
#define K_DIRECTOR 2186
#define K_ACTOR    8030
#define KP_GENRE    32
#define KP_DIRECTOR 2208
#define KP_ACTOR    8032
// half-element offsets inside the f16 weight workspace
#define WS_OFF_GENRE    0
#define WS_OFF_DIRECTOR (32 * KP_GENRE)
#define WS_OFF_ACTOR    (WS_OFF_DIRECTOR + 32 * KP_DIRECTOR)
#define WS_HALVES       (WS_OFF_ACTOR + 32 * KP_ACTOR)

union AFrag { v16h h; unsigned int u[8]; };

// ---- one-time f32 -> f16 weight conversion with zero padding to Kpad ----
__global__ void convert_w_kernel(const float* __restrict__ W, _Float16* __restrict__ out,
                                 int K, int Kpad) {
    int total = 32 * Kpad;
    for (int i = blockIdx.x * blockDim.x + threadIdx.x; i < total;
         i += gridDim.x * blockDim.x) {
        int r = i / Kpad, k = i - r * Kpad;
        out[i] = (k < K) ? (_Float16)W[r * K + k] : (_Float16)0.0f;
    }
}

// Mean-bag field for a 16-row tile via v_wmma_f32_16x16x32_f16.
//   out[16 x 32] = A[16 x K] * Wh^T[K x 32], rows scaled by 1/rowsum(A).
// Wh is (32 x Kpad) f16 row-major, zero-padded, so k >= K contributes 0.
__device__ __forceinline__ void bag_wmma(
    const int* __restrict__ x, const _Float16* __restrict__ Wh,
    int Kpad, int rowBase, int colStart, int K, int fieldOff,
    float* __restrict__ shH, int ldsRow0)
{
    const int lane = threadIdx.x & 31;
    const int half = lane >> 4;   // 0 or 1 (lane group)
    const int m    = lane & 15;   // A row / B column within tile

    const int* xrow = x + (size_t)(rowBase + m) * XCOLS + colStart;
    const _Float16* wr0 = Wh + (size_t)m * Kpad;          // N = 0..15
    const _Float16* wr1 = Wh + (size_t)(m + 16) * Kpad;   // N = 16..31

    v8f c0 = {};            // output cols 0..15
    v8f c1 = {};            // output cols 16..31
    unsigned int acc = 0;   // packed per-lane bag counts (lo/hi halves)

    int k0 = 0;
    // ---- clean steps: no bounds checks, 8-byte aligned pair loads ----
#pragma unroll 2
    for (; k0 + 32 <= K; k0 += 32) {
        __builtin_prefetch(xrow + k0 + 256, 0, 0);
        const int base = k0 + half * 16;
        AFrag a;
#pragma unroll
        for (int v = 0; v < 8; ++v) {
            // A k order used here (contiguous): lanes 0-15 K={0..15}, lanes 16-31 K={16..31}
            int ka = base + v * 2;
            int2 p = *(const int2*)(xrow + ka);
            unsigned int packed = (unsigned int)p.x | ((unsigned int)p.y << 16);
            acc += packed;                 // x in {0,1}: halves count entries
            a.u[v] = packed * 0x3C00u;     // 0 -> 0h, 1 -> 1.0h in each half
        }
        // B uses the same contiguous k order as A: VGPR v -> K = 2v (+16 for lanes 16-31), N = lane&15
        v16h b0 = *(const v16h*)(wr0 + k0 + half * 16);
        v16h b1 = *(const v16h*)(wr1 + k0 + half * 16);
        c0 = __builtin_amdgcn_wmma_f32_16x16x32_f16(false, a.h, false, b0, (short)0, c0, false, false);
        c1 = __builtin_amdgcn_wmma_f32_16x16x32_f16(false, a.h, false, b1, (short)0, c1, false, false);
    }
    // ---- masked tail step (also the entire genre field) ----
    if (k0 < K) {
        const int base = k0 + half * 16;
        AFrag a;
#pragma unroll
        for (int v = 0; v < 8; ++v) {
            int ka = base + v * 2;
            // reads stay inside the x row (padding < row end); mask to field range
            unsigned int x0 = (ka     < K) ? (unsigned int)xrow[ka]     : 0u;
            unsigned int x1 = (ka + 1 < K) ? (unsigned int)xrow[ka + 1] : 0u;
            unsigned int packed = x0 | (x1 << 16);
            acc += packed;
            a.u[v] = packed * 0x3C00u;
        }
        v16h b0 = *(const v16h*)(wr0 + k0 + half * 16);   // zero-padded to Kpad
        v16h b1 = *(const v16h*)(wr1 + k0 + half * 16);
        c0 = __builtin_amdgcn_wmma_f32_16x16x32_f16(false, a.h, false, b0, (short)0, c0, false, false);
        c1 = __builtin_amdgcn_wmma_f32_16x16x32_f16(false, a.h, false, b1, (short)0, c1, false, false);
    }

    // lane m holds K-subset {0..15,...} of row m; lane m+16 the complement
    float cnt = (float)((acc & 0xFFFFu) + (acc >> 16));
    cnt += __shfl_xor(cnt, 16, 32);
    float inv = 1.0f / cnt;

    // C layout: lane L, VGPR v -> M = v + 8*(L>=16), N = L&15
    float cv0[8], cv1[8];
#pragma unroll
    for (int v = 0; v < 8; ++v) { cv0[v] = c0[v]; cv1[v] = c1[v]; }
#pragma unroll
    for (int v = 0; v < 8; ++v) {
        int row = half * 8 + v;
        float s = __shfl(inv, row, 32);   // total count of 'row' lives in lane 'row'
        shH[(ldsRow0 + row) * HSTRIDE + fieldOff + m]      = cv0[v] * s;
        shH[(ldsRow0 + row) * HSTRIDE + fieldOff + 16 + m] = cv1[v] * s;
    }
}

__global__ __launch_bounds__(NTHREADS)
void upe_fused_kernel(
    const int*   __restrict__ x,
    const float* __restrict__ emb_rate,
    const _Float16* __restrict__ wsW,   // f16 weights: genre | director | actor
    const float* __restrict__ emb_gender,
    const float* __restrict__ emb_age,
    const float* __restrict__ emb_occ,
    const float* __restrict__ emb_area,
    const float* __restrict__ W1, const float* __restrict__ b1,
    const float* __restrict__ W2, const float* __restrict__ b2,
    const float* __restrict__ Wout, const float* __restrict__ bout,
    float* __restrict__ out)
{
    __shared__ float shH[ROWS_PER_BLOCK * HSTRIDE];   // fused feature rows h (64 x 256)
    __shared__ float sh1[ROWS_PER_BLOCK * 64];
    __shared__ float sh2[ROWS_PER_BLOCK * 64];

    const int blockRow0 = blockIdx.x * ROWS_PER_BLOCK;
    const int waveId    = threadIdx.x >> 5;
    const int ldsRow0   = waveId * 16;
    const int rowBase   = blockRow0 + ldsRow0;

    // ---- stage 1: three mean-bag GEMMs via WMMA (each wave owns 16 rows) ----
    bag_wmma(x, wsW + WS_OFF_GENRE,    KP_GENRE,    rowBase,    1, K_GENRE,    32, shH, ldsRow0);
    bag_wmma(x, wsW + WS_OFF_DIRECTOR, KP_DIRECTOR, rowBase,   26, K_DIRECTOR, 64, shH, ldsRow0);
    bag_wmma(x, wsW + WS_OFF_ACTOR,    KP_ACTOR,    rowBase, 2212, K_ACTOR,    96, shH, ldsRow0);

    // ---- stage 2: embedding gathers ----
    // h layout: [0:32) rate | [32:64) genre | [64:96) director | [96:128) actor
    //           [128:160) gender | [160:192) age | [192:224) occ | [224:256) area
    for (int i = threadIdx.x; i < ROWS_PER_BLOCK * 5 * 32; i += NTHREADS) {
        int r = i / 160;
        int j = i - r * 160;
        int e = j >> 5;
        int d = j & 31;
        const int* xr = x + (size_t)(blockRow0 + r) * XCOLS;
        const float* tab;
        int idx, coff;
        switch (e) {
            case 0:  idx = xr[0];     tab = emb_rate;   coff = 0;   break;
            case 1:  idx = xr[10242]; tab = emb_gender; coff = 128; break;
            case 2:  idx = xr[10243]; tab = emb_age;    coff = 160; break;
            case 3:  idx = xr[10244]; tab = emb_occ;    coff = 192; break;
            default: idx = xr[10245]; tab = emb_area;   coff = 224; break;
        }
        shH[r * HSTRIDE + coff + d] = tab[idx * 32 + d];
    }
    __syncthreads();

    // ---- stage 3: MLP layer 1 (256 -> 64, relu); 2 threads per row ----
    {
        int row   = threadIdx.x >> 1;
        int oBase = (threadIdx.x & 1) * 32;
        const float4* h4 = (const float4*)&shH[row * HSTRIDE];
        for (int o = oBase; o < oBase + 32; ++o) {
            const float4* w4 = (const float4*)&W1[o * 256];
            float acc = b1[o];
#pragma unroll 4
            for (int k = 0; k < 64; ++k) {
                float4 a = w4[k], b = h4[k];
                acc = fmaf(a.x, b.x, acc);
                acc = fmaf(a.y, b.y, acc);
                acc = fmaf(a.z, b.z, acc);
                acc = fmaf(a.w, b.w, acc);
            }
            sh1[row * 64 + o] = fmaxf(acc, 0.0f);
        }
    }
    __syncthreads();

    // ---- stage 4: MLP layer 2 (64 -> 64, relu) ----
    {
        int row   = threadIdx.x >> 1;
        int oBase = (threadIdx.x & 1) * 32;
        const float4* h4 = (const float4*)&sh1[row * 64];
        for (int o = oBase; o < oBase + 32; ++o) {
            const float4* w4 = (const float4*)&W2[o * 64];
            float acc = b2[o];
#pragma unroll
            for (int k = 0; k < 16; ++k) {
                float4 a = w4[k], b = h4[k];
                acc = fmaf(a.x, b.x, acc);
                acc = fmaf(a.y, b.y, acc);
                acc = fmaf(a.z, b.z, acc);
                acc = fmaf(a.w, b.w, acc);
            }
            sh2[row * 64 + o] = fmaxf(acc, 0.0f);
        }
    }
    __syncthreads();

    // ---- stage 5: output head (64 -> 1) ----
    if (threadIdx.x < ROWS_PER_BLOCK) {
        int row = threadIdx.x;
        const float4* h4 = (const float4*)&sh2[row * 64];
        const float4* w4 = (const float4*)Wout;
        float acc = bout[0];
#pragma unroll
        for (int k = 0; k < 16; ++k) {
            float4 a = w4[k], b = h4[k];
            acc = fmaf(a.x, b.x, acc);
            acc = fmaf(a.y, b.y, acc);
            acc = fmaf(a.z, b.z, acc);
            acc = fmaf(a.w, b.w, acc);
        }
        out[blockRow0 + row] = acc;
    }
}

extern "C" void kernel_launch(void* const* d_in, const int* in_sizes, int n_in,
                              void* d_out, int out_size, void* d_ws, size_t ws_size,
                              hipStream_t stream) {
    (void)in_sizes; (void)n_in; (void)ws_size; (void)out_size;

    const int*   x          = (const int*)  d_in[0];
    const float* emb_rate   = (const float*)d_in[1];
    const float* W_genre    = (const float*)d_in[2];
    const float* W_director = (const float*)d_in[3];
    const float* W_actor    = (const float*)d_in[4];
    const float* emb_gender = (const float*)d_in[5];
    const float* emb_age    = (const float*)d_in[6];
    const float* emb_occ    = (const float*)d_in[7];
    const float* emb_area   = (const float*)d_in[8];
    const float* W1         = (const float*)d_in[9];
    const float* b1         = (const float*)d_in[10];
    const float* W2         = (const float*)d_in[11];
    const float* b2         = (const float*)d_in[12];
    const float* Wout       = (const float*)d_in[13];
    const float* bout       = (const float*)d_in[14];
    float* out = (float*)d_out;
    _Float16* wsW = (_Float16*)d_ws;   // needs WS_HALVES*2 ~ 642 KB of scratch

    // stage 0: f32 -> f16 (zero-padded) weight staging, L2-resident afterwards
    hipLaunchKernelGGL(convert_w_kernel, dim3(64), dim3(256), 0, stream,
                       W_genre, wsW + WS_OFF_GENRE, K_GENRE, KP_GENRE);
    hipLaunchKernelGGL(convert_w_kernel, dim3(64), dim3(256), 0, stream,
                       W_director, wsW + WS_OFF_DIRECTOR, K_DIRECTOR, KP_DIRECTOR);
    hipLaunchKernelGGL(convert_w_kernel, dim3(64), dim3(256), 0, stream,
                       W_actor, wsW + WS_OFF_ACTOR, K_ACTOR, KP_ACTOR);

    const int B = 16384;
    hipLaunchKernelGGL(upe_fused_kernel, dim3(B / ROWS_PER_BLOCK), dim3(NTHREADS), 0, stream,
                       x, emb_rate, wsW, emb_gender, emb_age, emb_occ, emb_area,
                       W1, b1, W2, b2, Wout, bout, out);
}
